// QKVNormAttention_63496796504719
// MI455X (gfx1250) — matrix operand order repaired and checked by reading, not compiled, not run on hardware
//
#include <hip/hip_runtime.h>
#include <hip/hip_bf16.h>

// ---------------------------------------------------------------------------
// QKV-norm attention for MI455X (gfx1250), wave32 + v_wmma_f32_16x16x32_bf16.
// Compute-bound (~70 GFLOP vs ~50MB): WMMA GEMMs, fp32 accumulation, bf16
// operands. Every WMMA fragment is two ds_load_b128. Straight global->LDS
// tile copies use gfx1250 async-to-LDS (ASYNCcnt) with LDS double buffering;
// transposed tiles are software-pipelined through registers.
// ---------------------------------------------------------------------------

typedef __attribute__((ext_vector_type(16))) __bf16 v16bf;
typedef __attribute__((ext_vector_type(8)))  float  v8f;

union AF {                    // one WMMA bf16 operand fragment (16 bf16/lane)
  v16bf v;
  uint4          q[2];
  unsigned int   d[8];
  unsigned short u[16];
};
union U4 { uint4 v; unsigned short u[8]; };

#define SEQ   2048
#define HDIM  1024
#define NHEAD 16
#define HD    64
#define MROWS 4096            // B*S

__device__ __forceinline__ unsigned short f2bf(float f) {
  unsigned int u = __float_as_uint(f);
  u += 0x7FFFu + ((u >> 16) & 1u);          // round-to-nearest-even
  return (unsigned short)(u >> 16);
}

// ---- gfx1250 async global->LDS copy (ASYNCcnt), 16B per lane ----
__device__ __forceinline__ void async_ld_b128(const unsigned short* gptr,
                                              void* lptr) {
  unsigned laddr = (unsigned)(unsigned long long)lptr;   // low 32b = LDS offset
  asm volatile("global_load_async_to_lds_b128 %0, %1, off"
               :: "v"(laddr), "v"(gptr)
               : "memory");
}
__device__ __forceinline__ void wait_async0() {
  asm volatile("s_wait_asynccnt 0x0" ::: "memory");
}

// --------------------------- fp32 -> bf16 convert ---------------------------
__global__ __launch_bounds__(256) void cvt_bf16_kernel(
    const float* __restrict__ src, unsigned short* __restrict__ dst, int n) {
  int i = (blockIdx.x * 256 + threadIdx.x) * 4;
  if (i + 3 < n) {
    float4 f = *(const float4*)(src + i);
    union { unsigned short u[4]; uint2 v; } o;
    o.u[0] = f2bf(f.x); o.u[1] = f2bf(f.y);
    o.u[2] = f2bf(f.z); o.u[3] = f2bf(f.w);
    *(uint2*)(dst + i) = o.v;
  }
}

// ------------------- QKV projection GEMM + per-head LN ----------------------
// grid = (MROWS/64, HDIM/64, 3), block = 256 (8 waves).
__global__ __launch_bounds__(256) void qkv_gemm_ln_kernel(
    const unsigned short* __restrict__ xb,
    const unsigned short* __restrict__ wqb,
    const unsigned short* __restrict__ wkb,
    const unsigned short* __restrict__ wvb,
    const float* __restrict__ bq, const float* __restrict__ bk,
    const float* __restrict__ bv,
    const float* __restrict__ qsc, const float* __restrict__ qbi,
    const float* __restrict__ ksc, const float* __restrict__ kbi,
    const float* __restrict__ vsc, const float* __restrict__ vbi,
    unsigned short* __restrict__ qb,
    unsigned short* __restrict__ kb,
    unsigned short* __restrict__ vb) {
  __shared__ unsigned short As[2][64][40];  // A tile, double-buffered (async)
  __shared__ unsigned short Bt[64][40];     // B tile TRANSPOSED: [n][k]
  __shared__ float ct[64][65];
  __shared__ float lmean[64], lrstd[64];

  const unsigned short* W; const float* bias; const float* lsc; const float* lbi;
  unsigned short* out;
  if (blockIdx.z == 0)      { W = wqb; bias = bq; lsc = qsc; lbi = qbi; out = qb; }
  else if (blockIdx.z == 1) { W = wkb; bias = bk; lsc = ksc; lbi = kbi; out = kb; }
  else                      { W = wvb; bias = bv; lsc = vsc; lbi = vbi; out = vb; }

  const int tid = threadIdx.x, wave = tid >> 5, lane = tid & 31;
  const int row0 = blockIdx.x * 64, n0 = blockIdx.y * 64;
  const int mi = wave >> 1, njb = (wave & 1) * 2;
  const int ar = tid >> 2, ac = (tid & 3) * 8;     // A: 8 bf16 @ (ar, ac)
  const int br = tid >> 3, bc = (tid & 7) * 8;     // B: 8 bf16 @ (k=br, n=bc)

  v8f acc[2];
#pragma unroll
  for (int s = 0; s < 2; ++s)
#pragma unroll
    for (int r = 0; r < 8; ++r) acc[s][r] = 0.f;

  const unsigned short* aglb = xb + (size_t)(row0 + ar) * HDIM + ac;
  async_ld_b128(aglb, &As[0][ar][ac]);             // tile 0 -> LDS (async)
  U4 gbw; gbw.v = *(const uint4*)(W + (size_t)br * HDIM + n0 + bc);

  for (int k0 = 0; k0 < HDIM; k0 += 32) {
    const int buf = (k0 >> 5) & 1;
#pragma unroll
    for (int i = 0; i < 8; ++i) Bt[bc + i][br] = gbw.u[i];   // transpose store
    wait_async0();                                 // our A-tile writes landed
    __syncthreads();                               // everyone's landed

    if (k0 + 32 < HDIM) {                          // next tile into other buf
      async_ld_b128(aglb + k0 + 32, &As[buf ^ 1][ar][ac]);
      gbw.v = *(const uint4*)(W + (size_t)(k0 + 32 + br) * HDIM + n0 + bc);
    }

    AF a;                                          // A fragment: 2x b128
    {
      int arow = mi * 16 + (lane & 15);
      int kbse = (lane >> 4) * 8;
      a.q[0] = *(const uint4*)&As[buf][arow][kbse];
      a.q[1] = *(const uint4*)&As[buf][arow][16 + kbse];
    }
#pragma unroll
    for (int s = 0; s < 2; ++s) {                  // B fragment: 2x b128
      AF b;
      int col = (njb + s) * 16 + (lane & 15);
      int kb2 = (lane >> 4) * 8;
      b.q[0] = *(const uint4*)&Bt[col][kb2];
      b.q[1] = *(const uint4*)&Bt[col][kb2 + 16];
      acc[s] = __builtin_amdgcn_wmma_f32_16x16x32_bf16(
          false, a.v, false, b.v, (short)0, acc[s], false, false);
    }
    __syncthreads();
  }

  // epilogue: bias + stash into LDS
#pragma unroll
  for (int s = 0; s < 2; ++s) {
    int col = (njb + s) * 16 + (lane & 15);
    float bval = bias[n0 + col];
#pragma unroll
    for (int r = 0; r < 8; ++r) {
      int rw = mi * 16 + ((lane >> 4) << 3) + r;
      ct[rw][col] = acc[s][r] + bval;
    }
  }
  __syncthreads();

  if (tid < 64) {                                  // per-row LN stats (HD=64)
    float sum = 0.f;
    for (int c = 0; c < 64; ++c) sum += ct[tid][c];
    float mean = sum * (1.f / 64.f);
    float var = 0.f;
    for (int c = 0; c < 64; ++c) { float d = ct[tid][c] - mean; var += d * d; }
    lmean[tid] = mean;
    lrstd[tid] = rsqrtf(var * (1.f / 64.f) + 1e-6f);
  }
  __syncthreads();

  {
    int m = tid >> 2, cb = (tid & 3) * 16;
    float mean = lmean[m], rstd = lrstd[m];
    size_t base = (size_t)(row0 + m) * HDIM + n0;
    union { unsigned short u[16]; uint4 q[2]; } pk;
#pragma unroll
    for (int c2 = 0; c2 < 16; ++c2) {
      int c = cb + c2;
      float v = (ct[m][c] - mean) * rstd * lsc[c] + lbi[c];
      pk.u[c2] = f2bf(v);
    }
    *(uint4*)(out + base + cb)     = pk.q[0];
    *(uint4*)(out + base + cb + 8) = pk.q[1];
  }
}

// ---------------------- flash attention per (b,h,qtile) ---------------------
// grid = (SEQ/64, NHEAD, B), block = 256.
__global__ __launch_bounds__(256) void attn_kernel(
    const unsigned short* __restrict__ qb,
    const unsigned short* __restrict__ kbuf,
    const unsigned short* __restrict__ vbuf,
    unsigned short* __restrict__ ab) {
  __shared__ unsigned short Qs[64][72];      // [q][hd]  (async, resident)
  __shared__ unsigned short Ks[2][64][72];   // [key][hd] double-buf (async)
  __shared__ unsigned short Vt[64][72];      // V TRANSPOSED: [hd][key]
  __shared__ float st[64][65];
  __shared__ unsigned short pt[64][72];      // bf16 probs [q][key]
  __shared__ float rm[64], rl[64], ral[64], rmn[64];
  __shared__ float red[64][4];

  const int tid = threadIdx.x, wave = tid >> 5, lane = tid & 31;
  const int mi = wave >> 1, njb = (wave & 1) * 2;
  const size_t hoff  = (size_t)blockIdx.y * HD;
  const size_t brow0 = (size_t)blockIdx.z * SEQ;
  const size_t qrow0 = brow0 + blockIdx.x * 64;
  const int srow = tid >> 2, sq = tid & 3;   // softmax: 4 threads / row

  // issue Q tile + K tile 0 async; V tile 0 through registers
  U4 vreg[2];
#pragma unroll
  for (int it = 0; it < 2; ++it) {
    int e = tid * 2 + it, r = e >> 3, c = (e & 7) * 8;
    async_ld_b128(qb + (qrow0 + r) * HDIM + hoff + c, &Qs[r][c]);
    async_ld_b128(kbuf + (brow0 + r) * HDIM + hoff + c, &Ks[0][r][c]);
    vreg[it].v = *(const uint4*)(vbuf + (brow0 + r) * HDIM + hoff + c);
  }
  if (tid < 64) { rm[tid] = -1e30f; rl[tid] = 0.f; }

  v8f oacc[2];
#pragma unroll
  for (int s = 0; s < 2; ++s)
#pragma unroll
    for (int r = 0; r < 8; ++r) oacc[s][r] = 0.f;

  for (int kt = 0; kt < SEQ / 64; ++kt) {
    const int buf = kt & 1;
#pragma unroll
    for (int it = 0; it < 2; ++it) {         // commit V tile (transpose)
      int e = tid * 2 + it, r = e >> 3, c = (e & 7) * 8;
#pragma unroll
      for (int i = 0; i < 8; ++i) Vt[c + i][r] = vreg[it].u[i];
    }
    wait_async0();                           // our Q/K async writes landed
    __syncthreads();                         // all tiles in LDS

    if (kt + 1 < SEQ / 64) {                 // next K/V tile
      const size_t nrow0 = brow0 + (kt + 1) * 64;
#pragma unroll
      for (int it = 0; it < 2; ++it) {
        int e = tid * 2 + it, r = e >> 3, c = (e & 7) * 8;
        async_ld_b128(kbuf + (nrow0 + r) * HDIM + hoff + c,
                      &Ks[buf ^ 1][r][c]);
        vreg[it].v = *(const uint4*)(vbuf + (nrow0 + r) * HDIM + hoff + c);
      }
    }

    // ---- S = Q * K^T (HD=64 -> two K=32 WMMA steps) ----
    v8f sacc[2];
#pragma unroll
    for (int s = 0; s < 2; ++s)
#pragma unroll
      for (int r = 0; r < 8; ++r) sacc[s][r] = 0.f;

#pragma unroll
    for (int k0 = 0; k0 < HD; k0 += 32) {
      AF a;
      int arow = mi * 16 + (lane & 15);
      int kbse = (lane >> 4) * 8;
      a.q[0] = *(const uint4*)&Qs[arow][k0 + kbse];
      a.q[1] = *(const uint4*)&Qs[arow][k0 + 16 + kbse];
#pragma unroll
      for (int s = 0; s < 2; ++s) {
        AF b;                                // B = K^T: contiguous in Ks
        int key = (njb + s) * 16 + (lane & 15);
        int kb2 = k0 + (lane >> 4) * 8;
        b.q[0] = *(const uint4*)&Ks[buf][key][kb2];
        b.q[1] = *(const uint4*)&Ks[buf][key][kb2 + 16];
        sacc[s] = __builtin_amdgcn_wmma_f32_16x16x32_bf16(
            false, a.v, false, b.v, (short)0, sacc[s], false, false);
      }
    }

#pragma unroll
    for (int s = 0; s < 2; ++s) {            // scaled scores -> LDS
      int col = (njb + s) * 16 + (lane & 15);
#pragma unroll
      for (int r = 0; r < 8; ++r) {
        int rw = mi * 16 + ((lane >> 4) << 3) + r;
        st[rw][col] = sacc[s][r] * 0.125f;   // HD^-0.5
      }
    }
    __syncthreads();

    // ---- online softmax: 4 threads per row ----
    {
      float pmax = -1e30f;
#pragma unroll
      for (int i = 0; i < 16; ++i) pmax = fmaxf(pmax, st[srow][sq * 16 + i]);
      red[srow][sq] = pmax;
    }
    __syncthreads();
    if (tid < 64) {
      float mold = rm[tid];
      float mloc = fmaxf(fmaxf(red[tid][0], red[tid][1]),
                         fmaxf(red[tid][2], red[tid][3]));
      float mnew = fmaxf(mold, mloc);
      ral[tid] = __expf(mold - mnew);
      rmn[tid] = mnew;
      rm[tid]  = mnew;
    }
    __syncthreads();
    {
      float mnew = rmn[srow];
      union { unsigned short u[16]; uint4 q[2]; } pk;
      float sum = 0.f;
#pragma unroll
      for (int i = 0; i < 16; ++i) {
        float p = __expf(st[srow][sq * 16 + i] - mnew);
        sum += p;
        pk.u[i] = f2bf(p);
      }
      *(uint4*)&pt[srow][sq * 16]     = pk.q[0];
      *(uint4*)&pt[srow][sq * 16 + 8] = pk.q[1];
      red[srow][sq] = sum;
    }
    __syncthreads();
    if (tid < 64)
      rl[tid] = rl[tid] * ral[tid] +
                red[tid][0] + red[tid][1] + red[tid][2] + red[tid][3];
    __syncthreads();

    // ---- O = O*alpha + P*V ----
#pragma unroll
    for (int s = 0; s < 2; ++s)
#pragma unroll
      for (int r = 0; r < 8; ++r) {
        int rw = mi * 16 + ((lane >> 4) << 3) + r;
        oacc[s][r] *= ral[rw];
      }
#pragma unroll
    for (int k0 = 0; k0 < 64; k0 += 32) {
      AF a;                                  // A = probs (q x key)
      int arow = mi * 16 + (lane & 15);
      int kbse = (lane >> 4) * 8;
      a.q[0] = *(const uint4*)&pt[arow][k0 + kbse];
      a.q[1] = *(const uint4*)&pt[arow][k0 + 16 + kbse];
#pragma unroll
      for (int s = 0; s < 2; ++s) {
        AF b;                                // B = V: contiguous in Vt
        int col = (njb + s) * 16 + (lane & 15);
        int kb2 = k0 + (lane >> 4) * 8;
        b.q[0] = *(const uint4*)&Vt[col][kb2];
        b.q[1] = *(const uint4*)&Vt[col][kb2 + 16];
        oacc[s] = __builtin_amdgcn_wmma_f32_16x16x32_bf16(
            false, a.v, false, b.v, (short)0, oacc[s], false, false);
      }
    }
    __syncthreads();
  }

#pragma unroll
  for (int s = 0; s < 2; ++s) {              // O / l -> bf16
    int col = (njb + s) * 16 + (lane & 15);
#pragma unroll
    for (int r = 0; r < 8; ++r) {
      int rw = mi * 16 + ((lane >> 4) << 3) + r;
      float ov = oacc[s][r] / rl[rw];
      ab[(qrow0 + rw) * HDIM + hoff + col] = f2bf(ov);
    }
  }
}

// --------------------------- output projection ------------------------------
// grid = (MROWS/64, HDIM/64), block = 256. fp32 output to d_out.
__global__ __launch_bounds__(256) void out_proj_kernel(
    const unsigned short* __restrict__ ab,
    const unsigned short* __restrict__ wob,
    const float* __restrict__ bo,
    float* __restrict__ out) {
  __shared__ unsigned short As[2][64][40];
  __shared__ unsigned short Bt[64][40];

  const int tid = threadIdx.x, wave = tid >> 5, lane = tid & 31;
  const int row0 = blockIdx.x * 64, n0 = blockIdx.y * 64;
  const int mi = wave >> 1, njb = (wave & 1) * 2;
  const int ar = tid >> 2, ac = (tid & 3) * 8;
  const int br = tid >> 3, bc = (tid & 7) * 8;

  v8f acc[2];
#pragma unroll
  for (int s = 0; s < 2; ++s)
#pragma unroll
    for (int r = 0; r < 8; ++r) acc[s][r] = 0.f;

  const unsigned short* aglb = ab + (size_t)(row0 + ar) * HDIM + ac;
  async_ld_b128(aglb, &As[0][ar][ac]);
  U4 gbw; gbw.v = *(const uint4*)(wob + (size_t)br * HDIM + n0 + bc);

  for (int k0 = 0; k0 < HDIM; k0 += 32) {
    const int buf = (k0 >> 5) & 1;
#pragma unroll
    for (int i = 0; i < 8; ++i) Bt[bc + i][br] = gbw.u[i];
    wait_async0();
    __syncthreads();

    if (k0 + 32 < HDIM) {
      async_ld_b128(aglb + k0 + 32, &As[buf ^ 1][ar][ac]);
      gbw.v = *(const uint4*)(wob + (size_t)(k0 + 32 + br) * HDIM + n0 + bc);
    }

    AF a;
    {
      int arow = mi * 16 + (lane & 15);
      int kbse = (lane >> 4) * 8;
      a.q[0] = *(const uint4*)&As[buf][arow][kbse];
      a.q[1] = *(const uint4*)&As[buf][arow][16 + kbse];
    }
#pragma unroll
    for (int s = 0; s < 2; ++s) {
      AF b;
      int col = (njb + s) * 16 + (lane & 15);
      int kb2 = (lane >> 4) * 8;
      b.q[0] = *(const uint4*)&Bt[col][kb2];
      b.q[1] = *(const uint4*)&Bt[col][kb2 + 16];
      acc[s] = __builtin_amdgcn_wmma_f32_16x16x32_bf16(
          false, a.v, false, b.v, (short)0, acc[s], false, false);
    }
    __syncthreads();
  }

#pragma unroll
  for (int s = 0; s < 2; ++s) {
    int col = (njb + s) * 16 + (lane & 15);
    float bval = bo[n0 + col];
#pragma unroll
    for (int r = 0; r < 8; ++r) {
      int rw = mi * 16 + ((lane >> 4) << 3) + r;
      out[(size_t)(row0 + rw) * HDIM + n0 + col] = acc[s][r] + bval;
    }
  }
}

// ------------------------------- launcher -----------------------------------
extern "C" void kernel_launch(void* const* d_in, const int* in_sizes, int n_in,
                              void* d_out, int out_size, void* d_ws,
                              size_t ws_size, hipStream_t stream) {
  const float* x   = (const float*)d_in[0];
  const float* Wq  = (const float*)d_in[1];
  const float* bq  = (const float*)d_in[2];
  const float* Wk  = (const float*)d_in[3];
  const float* bk  = (const float*)d_in[4];
  const float* Wv  = (const float*)d_in[5];
  const float* bv  = (const float*)d_in[6];
  const float* qsc = (const float*)d_in[7];
  const float* qbi = (const float*)d_in[8];
  const float* ksc = (const float*)d_in[9];
  const float* kbi = (const float*)d_in[10];
  const float* vsc = (const float*)d_in[11];
  const float* vbi = (const float*)d_in[12];
  const float* Wo  = (const float*)d_in[13];
  const float* bo  = (const float*)d_in[14];
  float* out = (float*)d_out;

  char* w = (char*)d_ws;
  unsigned short* xb  = (unsigned short*)w; w += (size_t)MROWS * HDIM * 2;
  unsigned short* wqb = (unsigned short*)w; w += (size_t)HDIM * HDIM * 2;
  unsigned short* wkb = (unsigned short*)w; w += (size_t)HDIM * HDIM * 2;
  unsigned short* wvb = (unsigned short*)w; w += (size_t)HDIM * HDIM * 2;
  unsigned short* wob = (unsigned short*)w; w += (size_t)HDIM * HDIM * 2;
  unsigned short* qb  = (unsigned short*)w; w += (size_t)MROWS * HDIM * 2;
  unsigned short* kb2 = (unsigned short*)w; w += (size_t)MROWS * HDIM * 2;
  unsigned short* vb2 = (unsigned short*)w; w += (size_t)MROWS * HDIM * 2;
  unsigned short* ab  = (unsigned short*)w; w += (size_t)MROWS * HDIM * 2;

  const int nX = MROWS * HDIM;
  const int nW = HDIM * HDIM;
  cvt_bf16_kernel<<<nX / 1024, 256, 0, stream>>>(x,  xb,  nX);
  cvt_bf16_kernel<<<nW / 1024, 256, 0, stream>>>(Wq, wqb, nW);
  cvt_bf16_kernel<<<nW / 1024, 256, 0, stream>>>(Wk, wkb, nW);
  cvt_bf16_kernel<<<nW / 1024, 256, 0, stream>>>(Wv, wvb, nW);
  cvt_bf16_kernel<<<nW / 1024, 256, 0, stream>>>(Wo, wob, nW);

  qkv_gemm_ln_kernel<<<dim3(MROWS / 64, HDIM / 64, 3), 256, 0, stream>>>(
      xb, wqb, wkb, wvb, bq, bk, bv, qsc, qbi, ksc, kbi, vsc, vbi,
      qb, kb2, vb2);

  attn_kernel<<<dim3(SEQ / 64, NHEAD, 2), 256, 0, stream>>>(qb, kb2, vb2, ab);

  out_proj_kernel<<<dim3(MROWS / 64, HDIM / 64, 1), 256, 0, stream>>>(
      ab, wob, bo, out);
}
